// RNNLM_27333171872058
// MI455X (gfx1250) — matrix operand order repaired and checked by reading, not compiled, not run on hardware
//
#include <hip/hip_runtime.h>

#define VOCAB  32000
#define EMBED  32
#define HIDDEN 16
#define SEQ    128
#define BATCH  32
#define ROWS   (SEQ*BATCH)     // 4096
#define NTILE  (VOCAB/16)      // 2000 N-tiles of width 16
#define K2_WAVES 16
#define K2_TPB   (K2_WAVES*32) // 512

typedef __attribute__((ext_vector_type(16))) _Float16 v16h;
typedef __attribute__((ext_vector_type(8)))  float    v8f;

__device__ __forceinline__ float tanh_fast(float x) {
  // tanh(x) = 1 - 2/(exp(2x)+1); stable at +/-inf (exp->0 gives -1, exp->inf gives 1)
  float e = __expf(2.0f * x);
  return 1.0f - __fdividef(2.0f, e + 1.0f);
}

// ---------------------------------------------------------------------------
// K0: xw[t][b][j] = bias_x[j] + bias_h[j] + sum_e lookup[ids[t][b]][e]*Wx[e][j]
// Fully parallel; pulls the embedding gather + input GEMM off the serial path.
// ---------------------------------------------------------------------------
__global__ void __launch_bounds__(512) k0_embed_xw(
    const int*   __restrict__ ids,     // [SEQ*BATCH]
    const float* __restrict__ lookup,  // [VOCAB, EMBED]
    const float* __restrict__ Wx,      // [EMBED, HIDDEN]
    const float* __restrict__ bx,      // [HIDDEN]
    const float* __restrict__ bh,      // [HIDDEN]
    float*       __restrict__ xw)      // [SEQ*BATCH, HIDDEN]
{
  const int g  = blockIdx.x * 512 + threadIdx.x;   // 0 .. 65535
  const int j  = g & (HIDDEN - 1);
  const int tb = g >> 4;                           // 0 .. 4095
  const int idx = ids[tb];
  const float* xr = lookup + (size_t)idx * EMBED;
  float acc = bx[j] + bh[j];
#pragma unroll
  for (int e = 0; e < EMBED; ++e)
    acc += xr[e] * Wx[e * HIDDEN + j];
  xw[g] = acc;
}

// ---------------------------------------------------------------------------
// K1: sequential recurrence. One wave32; lane b owns batch row b entirely in
// registers (rows are independent) -> no barriers, no cross-lane ops.
// ---------------------------------------------------------------------------
__global__ void __launch_bounds__(32) k1_recurrence(
    const float* __restrict__ xw,     // [SEQ*BATCH, HIDDEN]
    const float* __restrict__ Wh,     // [HIDDEN, HIDDEN]
    const float* __restrict__ H0,     // [HIDDEN]
    float*       __restrict__ hAll)   // [SEQ*BATCH, HIDDEN]
{
  __shared__ __align__(16) float s_wh[HIDDEN * HIDDEN];
  const int b = threadIdx.x;
  for (int i = b; i < HIDDEN * HIDDEN; i += 32) s_wh[i] = Wh[i];
  __syncthreads();

  float h[HIDDEN];
#pragma unroll
  for (int j = 0; j < HIDDEN; ++j) h[j] = H0[j];

  // one-step register prefetch pipeline on xw (resides in L2 after K0)
  float4 n0, n1, n2, n3;
  {
    const float4* p = (const float4*)xw + (size_t)b * 4;
    n0 = p[0]; n1 = p[1]; n2 = p[2]; n3 = p[3];
  }

  for (int t = 0; t < SEQ; ++t) {
    float s[HIDDEN];
    s[0]=n0.x; s[1]=n0.y; s[2]=n0.z; s[3]=n0.w;
    s[4]=n1.x; s[5]=n1.y; s[6]=n1.z; s[7]=n1.w;
    s[8]=n2.x; s[9]=n2.y; s[10]=n2.z; s[11]=n2.w;
    s[12]=n3.x; s[13]=n3.y; s[14]=n3.z; s[15]=n3.w;

    if (t + 1 < SEQ) {
      const float4* p = (const float4*)xw + ((size_t)(t + 1) * BATCH + b) * 4;
      n0 = p[0]; n1 = p[1]; n2 = p[2]; n3 = p[3];
    }

#pragma unroll
    for (int k = 0; k < HIDDEN; ++k) {
      const float hk = h[k];
      const float4* wr = (const float4*)(s_wh + k * HIDDEN);
      const float4 w0 = wr[0], w1 = wr[1], w2 = wr[2], w3 = wr[3];
      s[0]  += hk * w0.x;  s[1]  += hk * w0.y;  s[2]  += hk * w0.z;  s[3]  += hk * w0.w;
      s[4]  += hk * w1.x;  s[5]  += hk * w1.y;  s[6]  += hk * w1.z;  s[7]  += hk * w1.w;
      s[8]  += hk * w2.x;  s[9]  += hk * w2.y;  s[10] += hk * w2.z;  s[11] += hk * w2.w;
      s[12] += hk * w3.x;  s[13] += hk * w3.y;  s[14] += hk * w3.z;  s[15] += hk * w3.w;
    }
#pragma unroll
    for (int j = 0; j < HIDDEN; ++j) h[j] = tanh_fast(s[j]);

    float4* op = (float4*)hAll + ((size_t)t * BATCH + b) * 4;
    op[0] = make_float4(h[0],  h[1],  h[2],  h[3]);
    op[1] = make_float4(h[4],  h[5],  h[6],  h[7]);
    op[2] = make_float4(h[8],  h[9],  h[10], h[11]);
    op[3] = make_float4(h[12], h[13], h[14], h[15]);
  }
}

// ---------------------------------------------------------------------------
// K2: logits GEMM via V_WMMA_F32_16X16X32_F16 + unstabilized log-softmax.
// Block m owns rows [16m,16m+16); 16 waves split V into 125 tiles each.
// Pass1 accumulates exp-sums; deterministic fixed-order LDS reduction;
// Pass2 recomputes the (cheap) WMMA and writes logp with NT stores.
// ---------------------------------------------------------------------------
__global__ void __launch_bounds__(K2_TPB) k2_logits(
    const float* __restrict__ hAll,   // [ROWS, HIDDEN]
    const float* __restrict__ Wo,     // [HIDDEN, VOCAB]
    const float* __restrict__ bo,     // [VOCAB]
    float*       __restrict__ out)    // [ROWS, VOCAB]
{
  __shared__ float red[K2_WAVES * 32 * 8];
  __shared__ float s_ls[16];

  const int tid    = threadIdx.x;
  const int wid    = tid >> 5;
  const int lane   = tid & 31;
  const int l16    = lane & 15;
  const bool hiH   = (lane >= 16);
  const int m      = blockIdx.x;   // 0..255
  const int r0     = m * 16;

  // ---- A operand: rows r0..r0+15, K=16 zero-padded to 32.
  // ISA layout: lanes 0-15 -> M=lane, elems 0..7 = K0..7, elems 8..15 = K16..23 (zero)
  //             lanes 16-31 -> M=lane-16, elems 0..7 = K8..15, elems 8..15 = K24..31 (zero)
  v16h a;
#pragma unroll
  for (int i = 0; i < 16; ++i) a[i] = (_Float16)0.0f;
  {
    const float* hrow = hAll + (size_t)(r0 + l16) * HIDDEN;
    const int koff = hiH ? 8 : 0;
#pragma unroll
    for (int i = 0; i < 8; ++i) a[i] = (_Float16)hrow[koff + i];
  }

  const int tilesPerWave = NTILE / K2_WAVES;  // 125
  const int t0 = wid * tilesPerWave;

  float acc[8];
#pragma unroll
  for (int i = 0; i < 8; ++i) acc[i] = 0.0f;

  // ---------------- pass 1: exp-sum ----------------
  for (int tt = 0; tt < tilesPerWave; ++tt) {
    const int n0 = (t0 + tt) * 16;
    // B operand: lane holds column N=l16; elems 0..7 = K(8h..8h+7), elems 8..15 = K16..31 (zero)
    v16h bmat;
#pragma unroll
    for (int i = 0; i < 16; ++i) bmat[i] = (_Float16)0.0f;
    {
      const float* wrow = Wo + (size_t)(hiH ? 8 : 0) * VOCAB + n0 + l16;
      __builtin_prefetch(wrow + 16, 0, 3);   // next tile for this wave
#pragma unroll
      for (int i = 0; i < 8; ++i) bmat[i] = (_Float16)wrow[(size_t)i * VOCAB];
    }
    v8f c = {};
    c = __builtin_amdgcn_wmma_f32_16x16x32_f16(false, a, false, bmat,
                                               (short)0, c, false, false);
    const float bv = bo[n0 + l16];
#pragma unroll
    for (int i = 0; i < 8; ++i) acc[i] += __expf(c[i] + bv);
  }

  // ---- deterministic block reduction (fixed order; no float atomics) ----
#pragma unroll
  for (int i = 0; i < 8; ++i) red[(wid * 32 + lane) * 8 + i] = acc[i];
  __syncthreads();
  if (tid < 16) {
    const int i  = tid & 7;               // c-register index
    const int lo = (tid < 8) ? 0 : 16;    // lane half contributing this row
    float s = 0.0f;
    for (int w = 0; w < K2_WAVES; ++w)
      for (int l = 0; l < 16; ++l)
        s += red[(w * 32 + lo + l) * 8 + i];
    s_ls[tid] = __logf(s);
  }
  __syncthreads();

  float ls[8];
  {
    const int base = hiH ? 8 : 0;
#pragma unroll
    for (int i = 0; i < 8; ++i) ls[i] = s_ls[base + i];
  }

  // ---------------- pass 2: recompute + write logp ----------------
  const int mbase = hiH ? 8 : 0;
  for (int tt = 0; tt < tilesPerWave; ++tt) {
    const int n0 = (t0 + tt) * 16;
    v16h bmat;
#pragma unroll
    for (int i = 0; i < 16; ++i) bmat[i] = (_Float16)0.0f;
    {
      const float* wrow = Wo + (size_t)(hiH ? 8 : 0) * VOCAB + n0 + l16;
#pragma unroll
      for (int i = 0; i < 8; ++i) bmat[i] = (_Float16)wrow[(size_t)i * VOCAB];
    }
    v8f c = {};
    c = __builtin_amdgcn_wmma_f32_16x16x32_f16(false, a, false, bmat,
                                               (short)0, c, false, false);
    const float bv = bo[n0 + l16];
#pragma unroll
    for (int i = 0; i < 8; ++i) {
      const size_t idx = (size_t)(r0 + mbase + i) * VOCAB + (size_t)(n0 + l16);
      // 524 MB streaming output: non-temporal so Wo stays resident in L2
      __builtin_nontemporal_store(c[i] + bv - ls[i], &out[idx]);
    }
  }
}

// ---------------------------------------------------------------------------
extern "C" void kernel_launch(void* const* d_in, const int* in_sizes, int n_in,
                              void* d_out, int out_size, void* d_ws, size_t ws_size,
                              hipStream_t stream) {
  const int*   ids    = (const int*)  d_in[0];  // [128,32] int32
  const float* lookup = (const float*)d_in[1];  // [32000,32]
  const float* Wx     = (const float*)d_in[2];  // [32,16]
  const float* Wh     = (const float*)d_in[3];  // [16,16]
  const float* Wo     = (const float*)d_in[4];  // [16,32000]
  const float* bx     = (const float*)d_in[5];  // [16]
  const float* bh     = (const float*)d_in[6];  // [16]
  const float* bo     = (const float*)d_in[7];  // [32000]
  const float* H0     = (const float*)d_in[8];  // [16]
  float* out = (float*)d_out;                   // [128,32,32000]

  float* wsf  = (float*)d_ws;
  float* xw   = wsf;                 // 65536 floats (256 KB)
  float* hAll = wsf + ROWS * HIDDEN; // 65536 floats (256 KB)

  k0_embed_xw  <<<128, 512, 0, stream>>>(ids, lookup, Wx, bx, bh, xw);
  k1_recurrence<<<1,    32, 0, stream>>>(xw, Wh, H0, hAll);
  k2_logits    <<<ROWS / 16, K2_TPB, 0, stream>>>(hAll, Wo, bo, out);
}
